// MultiHeadSlotAttentionImplicit_17514876634185
// MI455X (gfx1250) — matrix operand order repaired and checked by reading, not compiled, not run on hardware
//
#include <hip/hip_runtime.h>

#define B_  32
#define N_  4096
#define D_  256
#define H_  4
#define K_  8
#define T_  3
#define DH_ 64
#define SCALE_  (0.125f)   /* 64^-0.5 */
#define EPS_    (1e-8f)
#define LN_EPS_ (1e-5f)

typedef unsigned short ushort_t;
typedef unsigned int   uint_t;
typedef __attribute__((ext_vector_type(16))) __bf16          v16bf;
typedef __attribute__((ext_vector_type(8)))  float           v8f;
typedef __attribute__((ext_vector_type(16))) unsigned short  us16;
typedef __attribute__((ext_vector_type(8)))  unsigned short  us8;

__device__ __forceinline__ ushort_t f2bf(float f) {
  uint_t u = __builtin_bit_cast(uint_t, f);
  u += 0x7FFFu + ((u >> 16) & 1u);          // round-to-nearest-even
  return (ushort_t)(u >> 16);
}

__device__ __forceinline__ v8f wmma_bf16(us16 a, us16 b, v8f c) {
  return __builtin_amdgcn_wmma_f32_16x16x32_bf16(
      false, __builtin_bit_cast(v16bf, a),
      false, __builtin_bit_cast(v16bf, b),
      (short)0, c, false, false);
}

// ---------------------------------------------------------------------------
// fp32 -> bf16 weight conversion (w_k, w_v), once
// ---------------------------------------------------------------------------
__global__ void k_convert_w(const float* __restrict__ wk, const float* __restrict__ wv,
                            ushort_t* __restrict__ wkb, ushort_t* __restrict__ wvb) {
  int i = blockIdx.x * blockDim.x + threadIdx.x;
  if (i < D_ * D_) { wkb[i] = f2bf(wk[i]); wvb[i] = f2bf(wv[i]); }
}

// slots0 = broadcast(mu)
__global__ void k_init_slots(const float* __restrict__ mu, float* __restrict__ slots) {
  int i = blockIdx.x * blockDim.x + threadIdx.x;
  if (i < B_ * K_ * D_) slots[i] = mu[i % (K_ * D_)];
}

// ---------------------------------------------------------------------------
// Fused LN(x) + K/V projection, bf16 WMMA.
//   kf : [b][n][256]        (bf16, row-contiguous per token -> dots B-loads)
//   vT : [b][h][64][4096]   (bf16, n-contiguous per feature -> updates B-loads)
// grid = (B*N)/16 blocks of 128 threads (4 waves)
// ---------------------------------------------------------------------------
__global__ void __launch_bounds__(128) k_proj(
    const float* __restrict__ x, const float* __restrict__ g_in, const float* __restrict__ b_in,
    const ushort_t* __restrict__ wkb, const ushort_t* __restrict__ wvb,
    ushort_t* __restrict__ kf, ushort_t* __restrict__ vT) {
  __shared__ __align__(32) ushort_t xn[16][D_];
  __shared__ float rs[16][8], rq[16][8];
  __shared__ float mean_s[16], inv_s[16];
  const int tid = threadIdx.x;
  const int g0  = blockIdx.x * 16;

  // ---- LayerNorm of 16 rows (8 threads / row, 32 cols each) ----
  {
    int r = tid >> 3, seg = tid & 7;
    const float* row = x + (size_t)(g0 + r) * D_ + seg * 32;
    float v[32]; float s = 0.f, q = 0.f;
#pragma unroll
    for (int i = 0; i < 32; i++) { v[i] = row[i]; s += v[i]; q += v[i] * v[i]; }
    rs[r][seg] = s; rq[r][seg] = q;
    __syncthreads();
    if (tid < 16) {
      float S = 0.f, Q = 0.f;
      for (int i = 0; i < 8; i++) { S += rs[tid][i]; Q += rq[tid][i]; }
      float m = S * (1.0f / D_);
      mean_s[tid] = m;
      inv_s[tid]  = rsqrtf(Q * (1.0f / D_) - m * m + LN_EPS_);
    }
    __syncthreads();
    float m = mean_s[r], inv = inv_s[r];
#pragma unroll
    for (int i = 0; i < 32; i++) {
      int c = seg * 32 + i;
      xn[r][c] = f2bf((v[i] - m) * inv * g_in[c] + b_in[c]);
    }
    __syncthreads();
  }

  // ---- WMMA GEMM: 32 (matrix, colTile) pairs over 4 waves ----
  const int wave = tid >> 5, lane = tid & 31;
  const int l16 = lane & 15, half = lane >> 4;
  const int b = g0 / N_;
  const int n_base = g0 % N_;

  for (int p = 0; p < 8; p++) {
    int idx = wave * 8 + p;
    int mat = idx >> 4;         // 0 -> K, 1 -> V
    int ct  = idx & 15;         // column tile (16 cols each)
    const ushort_t* w = mat ? wvb : wkb;
    v8f acc = {};
#pragma unroll
    for (int kt = 0; kt < 8; kt++) {
      // A tile from LDS (contiguous K: 16 elems per lane at 16*half)
      us16 a = *reinterpret_cast<const us16*>(&xn[l16][kt * 32 + 16 * half]);
      // B tile: column (= output feature) per lane, contiguous K
      us16 bt = *reinterpret_cast<const us16*>(
          w + (size_t)(ct * 16 + l16) * D_ + kt * 32 + 16 * half);
      acc = wmma_bf16(a, bt, acc);
    }
    // C/D layout: col = l16, row = r + 8*half
    if (mat == 0) {
#pragma unroll
      for (int r = 0; r < 8; r++) {
        int m = r + 8 * half;
        kf[(size_t)(b * N_ + n_base + m) * D_ + ct * 16 + l16] = f2bf(acc[r]);
      }
    } else {
      int j = ct * 16 + l16;
      int h = j >> 6, dloc = j & 63;
#pragma unroll
      for (int r = 0; r < 8; r++) {
        int m = r + 8 * half;
        vT[((size_t)((b * H_ + h) * DH_ + dloc)) * N_ + (n_base + m)] = f2bf(acc[r]);
      }
    }
  }
}

// ---------------------------------------------------------------------------
// q = split(LN(slots) @ w_q.T)  ->  qb [b][h][k][64] bf16
// grid = B*K blocks of 256 threads (one thread per output column)
// ---------------------------------------------------------------------------
__global__ void __launch_bounds__(256) k_qproj(
    const float* __restrict__ slots, const float* __restrict__ g_slot,
    const float* __restrict__ b_slot, const float* __restrict__ wq,
    ushort_t* __restrict__ qb) {
  __shared__ float sh[D_];
  __shared__ float red[256];
  const int row = blockIdx.x;   // b*K + k
  const int col = threadIdx.x;
  float v = slots[(size_t)row * D_ + col];
  red[col] = v; __syncthreads();
  for (int s = 128; s > 0; s >>= 1) { if (col < s) red[col] += red[col + s]; __syncthreads(); }
  float m = red[0] * (1.0f / D_); __syncthreads();
  float d = v - m;
  red[col] = d * d; __syncthreads();
  for (int s = 128; s > 0; s >>= 1) { if (col < s) red[col] += red[col + s]; __syncthreads(); }
  float inv = rsqrtf(red[0] * (1.0f / D_) + LN_EPS_);
  sh[col] = d * inv * g_slot[col] + b_slot[col];
  __syncthreads();
  const float* wr = wq + (size_t)col * D_;
  float acc = 0.f;
  for (int i = 0; i < D_; i++) acc += sh[i] * wr[i];
  int b = row / K_, k = row % K_;
  int h = col >> 6, dloc = col & 63;
  qb[(((b * H_ + h) * K_ + k) * DH_) + dloc] = f2bf(acc);
}

// ---------------------------------------------------------------------------
// Fused attention step: dots -> softmax(slots) -> WMMA accumulate updates,
// renorm over N folded into final scale.  av : [b][k][256] fp32 (merged heads)
// grid = B*H blocks of 128 threads (4 waves)
// ---------------------------------------------------------------------------
__global__ void __launch_bounds__(128) k_attn(
    const ushort_t* __restrict__ qb, const ushort_t* __restrict__ kf,
    const ushort_t* __restrict__ vT, float* __restrict__ av) {
  __shared__ __align__(32) ushort_t attn_sh[4][16][32];   // per-wave A staging
  __shared__ float upd_sh[8][DH_];
  __shared__ float rowsum_sh[8];
  const int tid = threadIdx.x;
  const int wave = tid >> 5, lane = tid & 31;
  const int l16 = lane & 15, half = lane >> 4;
  const int bh = blockIdx.x;
  const int b = bh / H_, h = bh % H_;

  for (int i = tid; i < 4 * 16 * 32; i += 128) ((ushort_t*)attn_sh)[i] = 0;
  for (int i = tid; i < 8 * DH_; i += 128) ((float*)upd_sh)[i] = 0.f;
  if (tid < 8) rowsum_sh[tid] = 0.f;
  __syncthreads();

  // q A-tiles (kt = 0,1); rows 8..15 zero (slot padding)
  us16 qa[2];
  {
    const ushort_t* qrow = qb + (size_t)(b * H_ + h) * K_ * DH_;
#pragma unroll
    for (int kt = 0; kt < 2; kt++) {
      us16 a = {};
      if (l16 < K_) {
        const ushort_t* r = qrow + l16 * DH_ + kt * 32;
        a = *reinterpret_cast<const us16*>(r + 16 * half);
      }
      qa[kt] = a;
    }
  }

  v8f upd[4] = {v8f{}, v8f{}, v8f{}, v8f{}};
  float rsum[8];
#pragma unroll
  for (int r = 0; r < 8; r++) rsum[r] = 0.f;

  const ushort_t* kbase = kf + (size_t)b * N_ * D_ + h * DH_;
  const ushort_t* vbase = vT + (size_t)(b * H_ + h) * DH_ * N_;
  ushort_t* ash = &attn_sh[wave][0][0];

  for (int chunk = wave; chunk < N_ / 32; chunk += 4) {
    int n0 = chunk * 32;
#pragma unroll
    for (int sub = 0; sub < 2; sub++) {
      int nt = n0 + sub * 16;
      v8f acc = {};
#pragma unroll
      for (int kt = 0; kt < 2; kt++) {
        us16 bt = *reinterpret_cast<const us16*>(
            kbase + (size_t)(nt + l16) * D_ + kt * 32 + 16 * half);
        acc = wmma_bf16(qa[kt], bt, acc);
      }
      // softmax over the 8 slots: each column's logits live in one lane (half==0)
      float e[8]; float mx = -3.4e38f;
#pragma unroll
      for (int r = 0; r < 8; r++) { e[r] = acc[r] * SCALE_; mx = fmaxf(mx, e[r]); }
      float s = 0.f;
#pragma unroll
      for (int r = 0; r < 8; r++) { e[r] = __expf(e[r] - mx); s += e[r]; }
      float isv = 1.0f / s;
      if (half == 0) {
#pragma unroll
        for (int r = 0; r < 8; r++) {
          float a_ = e[r] * isv;
          rsum[r] += a_;
          ash[r * 32 + sub * 16 + l16] = f2bf(a_);
        }
      }
    }
    __syncthreads();
    // updates: A = attn tile [16 x 32] from LDS, B = vT (contiguous per lane)
    us16 at;
    {
      const ushort_t* arow = ash + l16 * 32;
      at = *reinterpret_cast<const us16*>(arow + 16 * half);
    }
#pragma unroll
    for (int ct = 0; ct < 4; ct++) {
      us16 bt = *reinterpret_cast<const us16*>(
          vbase + (size_t)(ct * 16 + l16) * N_ + n0 + 16 * half);
      upd[ct] = wmma_bf16(at, bt, upd[ct]);
    }
    __syncthreads();
  }

  // cross-wave reduction (valid rows 0..7 are in half==0 lanes)
  if (half == 0) {
#pragma unroll
    for (int r = 0; r < 8; r++) atomicAdd(&rowsum_sh[r], rsum[r]);
#pragma unroll
    for (int ct = 0; ct < 4; ct++)
#pragma unroll
      for (int r = 0; r < 8; r++)
        atomicAdd(&upd_sh[r][ct * 16 + l16], upd[ct][r]);
  }
  __syncthreads();
  for (int i = tid; i < 8 * DH_; i += 128) {
    int r = i / DH_, d = i % DH_;
    float val = upd_sh[r][d] / (rowsum_sh[r] + EPS_);
    av[((size_t)(b * K_ + r)) * D_ + h * DH_ + d] = val;
  }
}

// ---------------------------------------------------------------------------
// Per-slot chain: w_comb proj -> GRU -> LN -> MLP residual. 256 rows, scalar.
// grid = B*K blocks of 256 threads (one thread per column)
// ---------------------------------------------------------------------------
__global__ void __launch_bounds__(256) k_slot_update(
    const float* __restrict__ av, const float* __restrict__ slots_in,
    const float* __restrict__ w_comb,
    const float* __restrict__ w_ih, const float* __restrict__ w_hh,
    const float* __restrict__ b_ih, const float* __restrict__ b_hh,
    const float* __restrict__ g_mlp, const float* __restrict__ b_mlp,
    const float* __restrict__ w1, const float* __restrict__ b1,
    const float* __restrict__ w2, const float* __restrict__ b2,
    float* __restrict__ slots_out) {
  __shared__ float av_sh[D_], h_sh[D_], u_sh[D_], ln_sh[D_], hid_sh[D_];
  __shared__ float red[256];
  const int row = blockIdx.x, col = threadIdx.x;
  av_sh[col] = av[(size_t)row * D_ + col];
  h_sh[col]  = slots_in[(size_t)row * D_ + col];
  __syncthreads();
  // updates = av @ w_comb.T
  {
    const float* wr = w_comb + (size_t)col * D_;
    float a = 0.f;
    for (int i = 0; i < D_; i++) a += av_sh[i] * wr[i];
    u_sh[col] = a;
  }
  __syncthreads();
  // GRU (torch GRUCell semantics)
  float gi[3], gh[3];
#pragma unroll
  for (int g = 0; g < 3; g++) {
    const float* wi = w_ih + (size_t)(g * D_ + col) * D_;
    const float* wh = w_hh + (size_t)(g * D_ + col) * D_;
    float a = b_ih[g * D_ + col], c = b_hh[g * D_ + col];
    for (int i = 0; i < D_; i++) { a += u_sh[i] * wi[i]; c += h_sh[i] * wh[i]; }
    gi[g] = a; gh[g] = c;
  }
  float r_ = 1.0f / (1.0f + __expf(-(gi[0] + gh[0])));
  float z_ = 1.0f / (1.0f + __expf(-(gi[1] + gh[1])));
  float n_ = tanhf(gi[2] + r_ * gh[2]);
  float nh = (1.0f - z_) * n_ + z_ * h_sh[col];
  // LN(nh)
  red[col] = nh; __syncthreads();
  for (int s = 128; s > 0; s >>= 1) { if (col < s) red[col] += red[col + s]; __syncthreads(); }
  float m = red[0] * (1.0f / D_); __syncthreads();
  float d = nh - m;
  red[col] = d * d; __syncthreads();
  for (int s = 128; s > 0; s >>= 1) { if (col < s) red[col] += red[col + s]; __syncthreads(); }
  float inv = rsqrtf(red[0] * (1.0f / D_) + LN_EPS_);
  ln_sh[col] = d * inv * g_mlp[col] + b_mlp[col];
  __syncthreads();
  // hid = relu(ln @ w1.T + b1)
  {
    const float* wr = w1 + (size_t)col * D_;
    float a = b1[col];
    for (int i = 0; i < D_; i++) a += ln_sh[i] * wr[i];
    hid_sh[col] = fmaxf(a, 0.f);
  }
  __syncthreads();
  {
    const float* wr = w2 + (size_t)col * D_;
    float a = b2[col];
    for (int i = 0; i < D_; i++) a += hid_sh[i] * wr[i];
    slots_out[(size_t)row * D_ + col] = nh + a;
  }
}

// ---------------------------------------------------------------------------
extern "C" void kernel_launch(void* const* d_in, const int* in_sizes, int n_in,
                              void* d_out, int out_size, void* d_ws, size_t ws_size,
                              hipStream_t stream) {
  const float* x      = (const float*)d_in[0];
  const float* mu     = (const float*)d_in[1];
  const float* w_q    = (const float*)d_in[2];
  const float* w_k    = (const float*)d_in[3];
  const float* w_v    = (const float*)d_in[4];
  const float* w_comb = (const float*)d_in[5];
  const float* g_in   = (const float*)d_in[6];
  const float* b_in   = (const float*)d_in[7];
  const float* g_slot = (const float*)d_in[8];
  const float* b_slot = (const float*)d_in[9];
  const float* g_mlp  = (const float*)d_in[10];
  const float* b_mlp  = (const float*)d_in[11];
  const float* w_ih   = (const float*)d_in[12];
  const float* w_hh   = (const float*)d_in[13];
  const float* b_ih   = (const float*)d_in[14];
  const float* b_hh   = (const float*)d_in[15];
  const float* w1     = (const float*)d_in[16];
  const float* b1     = (const float*)d_in[17];
  const float* w2     = (const float*)d_in[18];
  const float* b2     = (const float*)d_in[19];
  float* out = (float*)d_out;

  char* ws = (char*)d_ws;
  ushort_t* wkb = (ushort_t*)ws; ws += (size_t)D_ * D_ * 2;
  ushort_t* wvb = (ushort_t*)ws; ws += (size_t)D_ * D_ * 2;
  ushort_t* kf  = (ushort_t*)ws; ws += (size_t)B_ * N_ * D_ * 2;
  ushort_t* vT  = (ushort_t*)ws; ws += (size_t)B_ * N_ * D_ * 2;
  ushort_t* qb  = (ushort_t*)ws; ws += (size_t)B_ * H_ * K_ * DH_ * 2;
  float* av     = (float*)ws;    ws += (size_t)B_ * K_ * D_ * 4;
  float* sA     = (float*)ws;    ws += (size_t)B_ * K_ * D_ * 4;
  float* sB     = (float*)ws;    ws += (size_t)B_ * K_ * D_ * 4;

  k_convert_w<<<(D_ * D_ + 255) / 256, 256, 0, stream>>>(w_k, w_v, wkb, wvb);
  k_init_slots<<<(B_ * K_ * D_ + 255) / 256, 256, 0, stream>>>(mu, sA);
  k_proj<<<(B_ * N_) / 16, 128, 0, stream>>>(x, g_in, b_in, wkb, wvb, kf, vT);

  const float* cur = sA;
  for (int t = 0; t < T_; t++) {
    float* nxt = (t == T_ - 1) ? out : ((t & 1) ? sA : sB);
    k_qproj<<<B_ * K_, 256, 0, stream>>>(cur, g_slot, b_slot, w_q, qb);
    k_attn<<<B_ * H_, 128, 0, stream>>>(qb, kf, vT, av);
    k_slot_update<<<B_ * K_, 256, 0, stream>>>(av, cur, w_comb, w_ih, w_hh,
                                               b_ih, b_hh, g_mlp, b_mlp,
                                               w1, b1, w2, b2, nxt);
    cur = nxt;
  }
}
